// SelfAttention_24103356465510
// MI455X (gfx1250) — matrix operand order repaired and checked by reading, not compiled, not run on hardware
//
#include <hip/hip_runtime.h>

// ---------------------------------------------------------------------------
// Self-attention forward on gfx1250 (MI455X): bf16 WMMA everywhere.
//   b=2, t=2048, K=1024, H=16, d=64
// Round 2: 32x64 wave tile (no accumulator spills), hoisted pointer
// arithmetic, global_prefetch on the k-stream.
// ---------------------------------------------------------------------------

typedef __attribute__((ext_vector_type(16))) __bf16 v16bf;
typedef __attribute__((ext_vector_type(8)))  __bf16 v8bf;
typedef __attribute__((ext_vector_type(8)))  float  v8f;

#define TDIM  2048
#define KDIM  1024
#define NH    16
#define HD    64
#define BATCH 2
#define MTOT  4096   // BATCH*TDIM

__device__ __forceinline__ unsigned short f2bf(float f) {
    unsigned int u = __float_as_uint(f);
    unsigned int r = (u + 0x7FFFu + ((u >> 16) & 1u)) >> 16;
    return (unsigned short)r;
}

// ---------------------------------------------------------------------------
// fp32 -> bf16 conversion (4 elems / thread)
// ---------------------------------------------------------------------------
__global__ void __launch_bounds__(256) k_f32_to_bf16(
        const float* __restrict__ in, unsigned short* __restrict__ out, int n) {
    int i = (blockIdx.x * blockDim.x + threadIdx.x) * 4;
    if (i + 3 < n) {
        float4 f = *(const float4*)(in + i);
        unsigned int lo = (unsigned int)f2bf(f.x) | ((unsigned int)f2bf(f.y) << 16);
        unsigned int hi = (unsigned int)f2bf(f.z) | ((unsigned int)f2bf(f.w) << 16);
        uint2 p; p.x = lo; p.y = hi;
        *(uint2*)(out + i) = p;
    }
}

// ---------------------------------------------------------------------------
// Generic bf16 WMMA GEMM:  C[m][n] = sum_k A[m][k] * BT[n][k]
//   A  : row-major M x Kd bf16, leading dim lda
//   BT : row-major N x Kd bf16 (B transposed), leading dim ldb
// Block: 256 threads = 8 waves; wave tile 32x64 (2x4 WMMA tiles);
//   waves arranged (8/WN) x WN -> block tile (8/WN)*32 x WN*64.
// MODE: 0 = f32 store (scores)
//       1 = bf16 row-major store (ctx)
//       2 = bf16 scatter [b,h,t,d] (Q,K)
//       3 = bf16 scatter [b,h,d,t] (V^T)
//       4 = f32 + bias (final output)
// ---------------------------------------------------------------------------
template<int MODE, int WN>
__global__ void __launch_bounds__(256, 1) k_gemm(
        const unsigned short* __restrict__ A, int lda,
        const unsigned short* __restrict__ BT, int ldb,
        void* __restrict__ Cout, int ldc, int Kd,
        const float* __restrict__ bias)
{
    const int lane  = threadIdx.x & 31;
    const int wave  = threadIdx.x >> 5;
    const int waveN = wave % WN;
    const int waveM = wave / WN;
    const int m0 = blockIdx.x * ((8 / WN) * 32) + waveM * 32;
    const int n0 = blockIdx.y * (WN * 64) + waveN * 64;
    const int lhalf = lane >> 4;   // 0 | 1 : which 16-lane half
    const int lmod  = lane & 15;

    v8f acc[2][4];
    const v8f vzero = {};
    #pragma unroll
    for (int i = 0; i < 2; i++)
        #pragma unroll
        for (int j = 0; j < 4; j++) acc[i][j] = vzero;

    // --- Hoisted per-lane pointers (advance by 32 elems each k-step). ---
    // A fragment (16x32 bf16): lane<16 row=lane  K={0..15}
    //                          lane>=16 row=lane-16 K={16..31}
    const unsigned short* pA0 = A + (size_t)(m0 + lmod) * lda + lhalf * 16;
    const unsigned short* pA1 = pA0 + 16 * (size_t)lda;
    // B fragment (32x16 bf16): lane holds column n=lmod, 16 consecutive K
    // (lhalf selects K 0..15 / 16..31). Column n of B == row n of BT.
    const unsigned short* pB0 = BT + (size_t)(n0 + lmod) * ldb + lhalf * 16;
    const unsigned short* pB1 = pB0 + 16 * (size_t)ldb;
    const unsigned short* pB2 = pB1 + 16 * (size_t)ldb;
    const unsigned short* pB3 = pB2 + 16 * (size_t)ldb;

    union AFrag { v16bf v; v8bf h[2]; };

    for (int k0 = 0; k0 < Kd; k0 += 32) {
        AFrag a0, a1;
        a0.h[0] = *(const v8bf*)(pA0);
        a0.h[1] = *(const v8bf*)(pA0 + 8);
        a1.h[0] = *(const v8bf*)(pA1);
        a1.h[1] = *(const v8bf*)(pA1 + 8);
        const v16bf b0 = *(const v16bf*)(pB0);
        const v16bf b1 = *(const v16bf*)(pB1);
        const v16bf b2 = *(const v16bf*)(pB2);
        const v16bf b3 = *(const v16bf*)(pB3);

        // prefetch next k-step (speculative; whole cachelines)
        __builtin_prefetch(pA0 + 32, 0, 3);
        __builtin_prefetch(pB0 + 32, 0, 3);

        acc[0][0] = __builtin_amdgcn_wmma_f32_16x16x32_bf16(
            false, a0.v, false, b0, (short)0, acc[0][0], false, false);
        acc[0][1] = __builtin_amdgcn_wmma_f32_16x16x32_bf16(
            false, a0.v, false, b1, (short)0, acc[0][1], false, false);
        acc[0][2] = __builtin_amdgcn_wmma_f32_16x16x32_bf16(
            false, a0.v, false, b2, (short)0, acc[0][2], false, false);
        acc[0][3] = __builtin_amdgcn_wmma_f32_16x16x32_bf16(
            false, a0.v, false, b3, (short)0, acc[0][3], false, false);
        acc[1][0] = __builtin_amdgcn_wmma_f32_16x16x32_bf16(
            false, a1.v, false, b0, (short)0, acc[1][0], false, false);
        acc[1][1] = __builtin_amdgcn_wmma_f32_16x16x32_bf16(
            false, a1.v, false, b1, (short)0, acc[1][1], false, false);
        acc[1][2] = __builtin_amdgcn_wmma_f32_16x16x32_bf16(
            false, a1.v, false, b2, (short)0, acc[1][2], false, false);
        acc[1][3] = __builtin_amdgcn_wmma_f32_16x16x32_bf16(
            false, a1.v, false, b3, (short)0, acc[1][3], false, false);

        pA0 += 32; pA1 += 32;
        pB0 += 32; pB1 += 32; pB2 += 32; pB3 += 32;
    }

    // ---- Epilogue. C/D layout: lane holds column N=lmod; rows
    // M = lhalf*8 + j across the 8 accumulator VGPRs.
    #pragma unroll
    for (int mt = 0; mt < 2; mt++) {
        #pragma unroll
        for (int nt = 0; nt < 4; nt++) {
            #pragma unroll
            for (int j = 0; j < 8; j++) {
                const int m = m0 + mt * 16 + lhalf * 8 + j;
                const int n = n0 + nt * 16 + lmod;
                const float val = acc[mt][nt][j];
                if (MODE == 0) {
                    ((float*)Cout)[(size_t)m * ldc + n] = val;
                } else if (MODE == 1) {
                    ((unsigned short*)Cout)[(size_t)m * ldc + n] = f2bf(val);
                } else if (MODE == 2) {          // [b,h,t,d]
                    const int bi = m >> 11, ti = m & 2047;
                    const int h = n >> 6, dd = n & 63;
                    const size_t addr = (((size_t)(bi * NH + h)) * TDIM + ti) * HD + dd;
                    ((unsigned short*)Cout)[addr] = f2bf(val);
                } else if (MODE == 3) {          // [b,h,d,t]
                    const int bi = m >> 11, ti = m & 2047;
                    const int h = n >> 6, dd = n & 63;
                    const size_t addr = (((size_t)(bi * NH + h)) * HD + dd) * TDIM + ti;
                    ((unsigned short*)Cout)[addr] = f2bf(val);
                } else {                          // 4: f32 + bias
                    ((float*)Cout)[(size_t)m * ldc + n] = val + bias[n];
                }
            }
        }
    }
}

// ---------------------------------------------------------------------------
// Row softmax over 2048 keys, with key-side padding mask and 1/sqrt(d) scale.
// One block (256 threads) per query row; S f32 in, P bf16 out.
// ---------------------------------------------------------------------------
__global__ void __launch_bounds__(256) k_softmax(
        const float* __restrict__ S, unsigned short* __restrict__ P,
        const int* __restrict__ pad, int bi)
{
    __shared__ float red[256];
    const int q   = blockIdx.x;
    const int tid = threadIdx.x;
    const float scale = 0.125f;            // 1/sqrt(64)
    const float NEG = -3.0e38f;

    const float* row = S + (size_t)q * TDIM;
    float vals[8];
    float mx = NEG;
    #pragma unroll
    for (int i = 0; i < 8; i++) {
        const int kk = tid + i * 256;
        float s = row[kk];
        s = (pad[bi * TDIM + kk] == 0) ? NEG : s * scale;
        vals[i] = s;
        mx = fmaxf(mx, s);
    }
    red[tid] = mx; __syncthreads();
    for (int off = 128; off > 0; off >>= 1) {
        if (tid < off) red[tid] = fmaxf(red[tid], red[tid + off]);
        __syncthreads();
    }
    mx = red[0]; __syncthreads();

    float sum = 0.f;
    #pragma unroll
    for (int i = 0; i < 8; i++) {
        vals[i] = __expf(vals[i] - mx);
        sum += vals[i];
    }
    red[tid] = sum; __syncthreads();
    for (int off = 128; off > 0; off >>= 1) {
        if (tid < off) red[tid] += red[tid + off];
        __syncthreads();
    }
    const float inv = 1.f / red[0];

    unsigned short* prow = P + (size_t)q * TDIM;
    #pragma unroll
    for (int i = 0; i < 8; i++)
        prow[tid + i * 256] = f2bf(vals[i] * inv);
}

// ---------------------------------------------------------------------------
// Launch
// ---------------------------------------------------------------------------
extern "C" void kernel_launch(void* const* d_in, const int* in_sizes, int n_in,
                              void* d_out, int out_size, void* d_ws, size_t ws_size,
                              hipStream_t stream)
{
    const float* x       = (const float*)d_in[0];
    const int*   padding = (const int*)  d_in[1];
    const float* Wq      = (const float*)d_in[2];
    const float* Wk      = (const float*)d_in[3];
    const float* Wv      = (const float*)d_in[4];
    const float* Wo      = (const float*)d_in[5];
    const float* bo      = (const float*)d_in[6];
    float* out = (float*)d_out;

    // Workspace layout (bf16 stored as u16):
    unsigned short* XB  = (unsigned short*)d_ws;                 // [4096][1024]
    unsigned short* WQB = XB  + (size_t)MTOT * KDIM;             // [1024][1024]
    unsigned short* WKB = WQB + (size_t)KDIM * KDIM;
    unsigned short* WVB = WKB + (size_t)KDIM * KDIM;
    unsigned short* WOB = WVB + (size_t)KDIM * KDIM;
    unsigned short* Qb  = WOB + (size_t)KDIM * KDIM;             // [b,h,t,d]
    unsigned short* Kb  = Qb  + (size_t)MTOT * KDIM;             // [b,h,t,d]
    unsigned short* VTb = Kb  + (size_t)MTOT * KDIM;             // [b,h,d,t]
    unsigned short* CTX = VTb + (size_t)MTOT * KDIM;             // [4096][1024]
    float*          Sb  = (float*)(CTX + (size_t)MTOT * KDIM);   // [2048][2048] f32
    unsigned short* Pb  = (unsigned short*)(Sb + (size_t)TDIM * TDIM); // [2048][2048]

    // 1) Convert inputs to bf16
    k_f32_to_bf16<<<(MTOT * KDIM) / 1024, 256, 0, stream>>>(x,  XB,  MTOT * KDIM);
    k_f32_to_bf16<<<(KDIM * KDIM) / 1024, 256, 0, stream>>>(Wq, WQB, KDIM * KDIM);
    k_f32_to_bf16<<<(KDIM * KDIM) / 1024, 256, 0, stream>>>(Wk, WKB, KDIM * KDIM);
    k_f32_to_bf16<<<(KDIM * KDIM) / 1024, 256, 0, stream>>>(Wv, WVB, KDIM * KDIM);
    k_f32_to_bf16<<<(KDIM * KDIM) / 1024, 256, 0, stream>>>(Wo, WOB, KDIM * KDIM);

    // 2) Projections: M=4096, N=1024, Kd=1024. WN=2 -> block tile 128x128.
    k_gemm<2, 2><<<dim3(32, 8), 256, 0, stream>>>(XB, KDIM, WQB, KDIM, Qb,  0, KDIM, nullptr);
    k_gemm<2, 2><<<dim3(32, 8), 256, 0, stream>>>(XB, KDIM, WKB, KDIM, Kb,  0, KDIM, nullptr);
    k_gemm<3, 2><<<dim3(32, 8), 256, 0, stream>>>(XB, KDIM, WVB, KDIM, VTb, 0, KDIM, nullptr);

    // 3) Attention per (b,h)
    for (int bh = 0; bh < BATCH * NH; bh++) {
        const int bi = bh >> 4;
        const int h  = bh & 15;
        const unsigned short* Qh = Qb  + (size_t)bh * TDIM * HD;
        const unsigned short* Kh = Kb  + (size_t)bh * TDIM * HD;
        const unsigned short* Vh = VTb + (size_t)bh * HD * TDIM;

        // S = Q K^T : M=2048, N=2048, Kd=64. block tile 128x128.
        k_gemm<0, 2><<<dim3(16, 16), 256, 0, stream>>>(Qh, HD, Kh, HD, Sb, TDIM, HD, nullptr);

        // masked/scaled softmax -> P (bf16)
        k_softmax<<<TDIM, 256, 0, stream>>>(Sb, Pb, padding, bi);

        // ctx = P V : M=2048, N=64, Kd=2048. WN=1 -> block tile 256x64.
        unsigned short* ctxh = CTX + (size_t)bi * TDIM * KDIM + (size_t)h * HD;
        k_gemm<1, 1><<<dim3(8, 1), 256, 0, stream>>>(Pb, TDIM, Vh, TDIM, ctxh, KDIM, TDIM, nullptr);
    }

    // 4) Output projection + bias: M=4096, N=1024, Kd=1024, f32 out.
    k_gemm<4, 2><<<dim3(32, 8), 256, 0, stream>>>(CTX, KDIM, WOB, KDIM, out, KDIM, KDIM, bo);
}
